// Local_Ar_71760313582101
// MI455X (gfx1250) — matrix operand order repaired and checked by reading, not compiled, not run on hardware
//
#include <hip/hip_runtime.h>
#include <cmath>

// ---------------------------------------------------------------------------
// CDNA5 (gfx1250) implementation. All heavy math runs through
// v_wmma_f32_16x16x32_f16 (f16 A/B, f32 accumulate) via an LDS-staged GEMM.
// LDS tiles are filled with GLOBAL_LOAD_ASYNC_TO_LDS_B128 (ASYNCcnt path).
// Activations are kept NHWC f32; GEMM inputs are f16 (im2col / packed).
// Param flattening assumption: jax pytree order (dict keys sorted, lists in
// order) -> 1960 leaf arrays after x, sr.
// ---------------------------------------------------------------------------

typedef _Float16 v16h __attribute__((ext_vector_type(16)));
typedef _Float16 h8   __attribute__((ext_vector_type(8)));
typedef float    v8f  __attribute__((ext_vector_type(8)));

#define LDS_STR 40            // LDS row stride in halves (32 data + 8 pad)
#define CHUNK   8192          // M-chunk for im2col buffer

__device__ __forceinline__ float eluf(float v) {
  return v > 0.f ? v : (expf(v) - 1.f);
}

// async copy of 16 bytes: global -> LDS (per-lane addresses), ASYNCcnt
__device__ __forceinline__ void async_g2l_b128(_Float16* lds, const _Float16* g) {
  unsigned loff = (unsigned)(unsigned long long)(lds);  // addrspace(3) offset
  asm volatile("global_load_async_to_lds_b128 %0, %1, off"
               :: "v"(loff), "v"(g) : "memory");
}
__device__ __forceinline__ void async_g2l_b128_o16(_Float16* lds, const _Float16* g) {
  unsigned loff = (unsigned)(unsigned long long)(lds);
  asm volatile("global_load_async_to_lds_b128 %0, %1, off offset:16"
               :: "v"(loff), "v"(g) : "memory");
}
__device__ __forceinline__ void wait_async0() {
  asm volatile("s_wait_asynccnt 0x0" ::: "memory");
}

// ---------------------------------------------------------------------------
// WMMA GEMM:  D[M][N] = act( (A[M][Kp] x Bw[N][Kp]^T + bias)*scale+shift +res )
// Block tile 128(M) x 64(N), 8 waves, each wave 32x32 (2x2 wmma frags).
// ---------------------------------------------------------------------------
__global__ __launch_bounds__(256) void k_gemm(
    const _Float16* __restrict__ A, const _Float16* __restrict__ Bw,
    const float* __restrict__ bias, const float* __restrict__ scale,
    const float* __restrict__ shift, const float* __restrict__ resid, int ldr,
    float* __restrict__ outF, _Float16* __restrict__ outH, int ldo,
    int Kp, int act) {
  __shared__ _Float16 sA[128 * LDS_STR];
  __shared__ _Float16 sB[64 * LDS_STR];

  const int tid  = threadIdx.x;
  const int lane = tid & 31;
  const int wave = tid >> 5;
  const int m0 = blockIdx.x * 128;
  const int n0 = blockIdx.y * 64;
  const int mb = (wave & 3) * 32;   // wave M sub-tile
  const int nb = (wave >> 2) * 32;  // wave N sub-tile

  v8f acc[2][2];
#pragma unroll
  for (int i = 0; i < 2; i++)
#pragma unroll
    for (int j = 0; j < 2; j++) acc[i][j] = (v8f){0.f,0.f,0.f,0.f,0.f,0.f,0.f,0.f};

  const int arow = tid >> 1, ahalf = tid & 1;   // A: 128 rows x 2 halves(16h)
  const int brow = tid >> 2, bq = tid & 3;      // B: 64 rows x 4 chunks(8h)

  for (int k0 = 0; k0 < Kp; k0 += 32) {
    // stage tiles via async global->LDS (no VGPR bounce)
    async_g2l_b128(sA + arow * LDS_STR + ahalf * 16,
                   A + (size_t)(m0 + arow) * Kp + k0 + ahalf * 16);
    async_g2l_b128_o16(sA + arow * LDS_STR + ahalf * 16,
                       A + (size_t)(m0 + arow) * Kp + k0 + ahalf * 16);
    async_g2l_b128(sB + brow * LDS_STR + bq * 8,
                   Bw + (size_t)(n0 + brow) * Kp + k0 + bq * 8);
    wait_async0();      // own wave's LDS writes complete
    __syncthreads();    // all waves' writes visible

    v16h af[2], bf[2];
    const int koff = (lane >> 4) * 8;
#pragma unroll
    for (int mi = 0; mi < 2; mi++) {
      const _Float16* base = sA + (mb + mi * 16 + (lane & 15)) * LDS_STR + koff;
      union { v16h v; h8 h[2]; } u;
      u.h[0] = *(const h8*)(base);
      u.h[1] = *(const h8*)(base + 16);
      af[mi] = u.v;
    }
    const int kb = (lane >> 4) * 8;
#pragma unroll
    for (int ni = 0; ni < 2; ni++) {
      const _Float16* base = sB + (nb + ni * 16 + (lane & 15)) * LDS_STR + kb;
      union { v16h v; h8 h[2]; } u;
      u.h[0] = *(const h8*)(base);
      u.h[1] = *(const h8*)(base + 16);
      bf[ni] = u.v;
    }

#pragma unroll
    for (int mi = 0; mi < 2; mi++)
#pragma unroll
      for (int ni = 0; ni < 2; ni++)
        acc[mi][ni] = __builtin_amdgcn_wmma_f32_16x16x32_f16(
            false, af[mi], false, bf[ni], (short)0, acc[mi][ni], false, false);
    __syncthreads();
  }

  // Epilogue. D layout: vgpr r -> row r + (lane>>4)*8 ; col = lane&15
  // Hoist per-column affine terms: v' = v*g + (bias*g + sh)
  const int mrow = (lane >> 4) * 8;
  const int ncol = lane & 15;
  int nc[2];
  nc[0] = n0 + nb + ncol;
  nc[1] = n0 + nb + 16 + ncol;
  float ga[2] = {1.f, 1.f}, ba[2] = {0.f, 0.f};
  if (bias) { ba[0] = bias[nc[0]]; ba[1] = bias[nc[1]]; }
  if (scale) {
    ga[0] = scale[nc[0]]; ga[1] = scale[nc[1]];
    ba[0] = ba[0] * ga[0] + shift[nc[0]];
    ba[1] = ba[1] * ga[1] + shift[nc[1]];
  }
#pragma unroll
  for (int mi = 0; mi < 2; mi++)
#pragma unroll
    for (int ni = 0; ni < 2; ni++)
#pragma unroll
      for (int r = 0; r < 8; r++) {
        int m = m0 + mb + mi * 16 + mrow + r;
        int n = nc[ni];
        float v = acc[mi][ni][r] * ga[ni] + ba[ni];
        if (resid) v += resid[(size_t)m * ldr + n];
        if (act)   v = eluf(v);
        if (outF)  outF[(size_t)m * ldo + n] = v;
        if (outH)  outH[(size_t)m * ldo + n] = (_Float16)v;
      }
}

// ---------------------------------------------------------------------------
// Elementwise / data-movement kernels (grid-stride; n = total elements)
// ---------------------------------------------------------------------------
#define GS(i, n)                                                          \
  for (long long i = blockIdx.x * (long long)blockDim.x + threadIdx.x;    \
       i < (n); i += (long long)gridDim.x * blockDim.x)

__global__ void k_nchw2nhwc(const float* __restrict__ s, float* __restrict__ d,
                            int C, int HW, long long n) {
  GS(i, n) {
    long long c = i % C, t = i / C;
    long long p = t % HW, b = t / HW;
    d[i] = s[(b * C + c) * HW + p];
  }
}

// im2col to f16, K ordering k = (ky*KW+kx)*C + ci (matches k_wconv)
__global__ void k_im2col(const float* __restrict__ src, _Float16* __restrict__ dst,
                         int H, int W, int C, int KH, int KW, int strd, int pad,
                         int OH, int OW, int K, int Kp, int mOff, long long n) {
  GS(i, n) {
    int k = (int)(i % Kp);
    int mi = (int)(i / Kp);
    int m = mOff + mi;
    float v = 0.f;
    if (k < K) {
      int ci = k % C, p2 = k / C;
      int kx = p2 % KW, ky = p2 / KW;
      int ox = m % OW; int t = m / OW;
      int oy = t % OH; int b = t / OH;
      int y = oy * strd - pad + ky;
      int x = ox * strd - pad + kx;
      if (y >= 0 && y < H && x >= 0 && x < W)
        v = src[(((size_t)b * H + y) * W + x) * C + ci];
    }
    dst[(size_t)mi * Kp + k] = (_Float16)v;
  }
}

// conv weight OIHW -> f16 [O][Kp], k = (ky*KW+kx)*I + ci, zero pad k>=K
__global__ void k_wconv(const float* __restrict__ w, _Float16* __restrict__ dst,
                        int O, int I, int KH, int KW, int Kp, long long n) {
  GS(i, n) {
    int k = (int)(i % Kp);
    int o = (int)(i / Kp);
    float v = 0.f;
    int K = I * KH * KW;
    if (k < K) {
      int ci = k % I, p2 = k / I;
      int kx = p2 % KW, ky = p2 / KW;
      v = w[(((size_t)o * I + ci) * KH + ky) * KW + kx];
    }
    dst[i] = (_Float16)v;
  }
}

// linear weight (K,N) -> f16 [N][Kp] (transpose), zero pad
__global__ void k_wlin(const float* __restrict__ w, _Float16* __restrict__ dst,
                       int K, int N, int Kp, long long n) {
  GS(i, n) {
    int k = (int)(i % Kp);
    int nn = (int)(i / Kp);
    dst[i] = (_Float16)(k < K ? w[(size_t)k * N + nn] : 0.f);
  }
}

__global__ void k_upsample2(const float* __restrict__ s, float* __restrict__ d,
                            int H, int W, int C, long long n) {
  GS(i, n) {
    int c = (int)(i % C); long long t = i / C;
    int X = (int)(t % (2 * W)); t /= (2 * W);
    int Y = (int)(t % (2 * H)); long long b = t / (2 * H);
    d[i] = s[(((size_t)b * H + (Y >> 1)) * W + (X >> 1)) * C + c];
  }
}

// one-hot @ W == row gather; fused elu; rows (b,j,y,x), cols 64
__global__ void k_xe_gather(const int* __restrict__ xi, const float* __restrict__ b0,
                            const float* __restrict__ w, _Float16* __restrict__ dst,
                            long long n) {
  GS(i, n) {
    int c = (int)(i & 63); int m = (int)(i >> 6);
    int x4 = m & 15, y = (m >> 4) & 15, j = (m >> 8) & 15, bb = m >> 12;
    int Y = y * 4 + (j >> 2), X = x4 * 4 + (j & 3);
    int xv = xi[(bb * 64 + Y) * 64 + X];
    dst[i] = (_Float16)eluf(w[(size_t)xv * 64 + c] + b0[c]);
  }
}

// unfold(srf).transpose -> f16 rows (b,j,y,x) at column offset co, ld
__global__ void k_srf_gather(const float* __restrict__ srf, _Float16* __restrict__ dst,
                             int ld, int co, long long n) {
  GS(i, n) {
    int c = (int)(i & 63); int m = (int)(i >> 6);
    int x4 = m & 15, y = (m >> 4) & 15, j = (m >> 8) & 15, bb = m >> 12;
    int Y = y * 4 + (j >> 2), X = x4 * 4 + (j & 3);
    dst[(size_t)m * ld + co + c] = (_Float16)srf[((size_t)(bb * 64 + Y) * 64 + X) * 64 + c];
  }
}

// S[j][(b,y,x)][c]: j==0 -> init_feature ; j>0 -> agg rows (b,j-1,y,x)
__global__ void k_build_state(const float* __restrict__ initf, const float* __restrict__ agg,
                              float* __restrict__ S, long long n) {
  GS(i, n) {
    int c = (int)(i & 63); long long t = i >> 6;
    int m2 = (int)(t & 1023); int j = (int)(t >> 10);
    float v;
    if (j == 0) v = initf[(size_t)m2 * 64 + c];
    else {
      int x4 = m2 & 15, y = (m2 >> 4) & 15, bb = m2 >> 8;
      int row = ((bb * 16 + (j - 1)) * 16 + y) * 16 + x4;
      v = agg[(size_t)row * 64 + c];
    }
    S[i] = v;
  }
}

// gate h-conv input: dst[m2][c*l + j] = R[j*1024+m2][c]  (f16)
__global__ void k_pack_h(const float* __restrict__ R, _Float16* __restrict__ dst,
                         int l, long long n) {
  GS(i, n) {
    int c = (int)(i & 63); long long t = i >> 6;
    int j = (int)(t % l); int m2 = (int)(t / l);
    dst[(size_t)m2 * (l * 64) + c * l + j] = (_Float16)R[((size_t)j * 1024 + m2) * 64 + c];
  }
}

// final state -> rows (b,j,y,x) f16 at col offset co
__global__ void k_state_rows(const float* __restrict__ S, _Float16* __restrict__ dst,
                             int ld, int co, long long n) {
  GS(i, n) {
    int c = (int)(i & 63); int m = (int)(i >> 6);
    int x4 = m & 15, y = (m >> 4) & 15, j = (m >> 8) & 15, bb = m >> 12;
    int m2 = (bb * 16 + y) * 16 + x4;
    dst[(size_t)m * ld + co + c] = (_Float16)S[((size_t)j * 1024 + m2) * 64 + c];
  }
}

// fold: out[b,e,Y,X] = rows[(b, j=sy*4+sx, y, x)][e]
__global__ void k_fold_out(const float* __restrict__ rows, float* __restrict__ out,
                           long long n) {
  GS(i, n) {
    int X = (int)(i & 63); long long t = i >> 6;
    int Y = (int)(t & 63); t >>= 6;
    int e = (int)(t & 127); int bb = (int)(t >> 7);
    int y = Y >> 2, sy = Y & 3, x4 = X >> 2, sx = X & 3;
    int j = sy * 4 + sx;
    int m = ((bb * 16 + j) * 16 + y) * 16 + x4;
    out[i] = rows[(size_t)m * 128 + e];
  }
}

// ---------------------------------------------------------------------------
// Host orchestration
// ---------------------------------------------------------------------------
#define EW(KERN, N, ...)                                                     \
  do {                                                                       \
    long long _n = (N);                                                      \
    int _g = (int)((_n + 255) / 256);                                        \
    if (_g > 65535) _g = 65535;                                              \
    KERN<<<dim3(_g), dim3(256), 0, stream>>>(__VA_ARGS__, _n);               \
  } while (0)

extern "C" void kernel_launch(void* const* d_in, const int* in_sizes, int n_in,
                              void* d_out, int out_size, void* d_ws, size_t ws_size,
                              hipStream_t stream) {
  (void)in_sizes; (void)n_in; (void)out_size; (void)ws_size;
  const int* xin = (const int*)d_in[0];
  const float* sr = (const float*)d_in[1];
  auto F = [&](int i) { return (const float*)d_in[i]; };

  // param leaf indices (jax pytree order: dict keys sorted, lists in order)
  const int AGG = 2, GATES = 8, INIT0 = 1928, I1X1 = 1930, IDS = 1934,
            POUT = 1938, SRD = 1944, UNET = 1946, XE = 1956;
  // unet sorted: down(b,w)=+0,+1  e1=+2,+3  mid=+4,+5  out=+6,+7  up=+8,+9

  // workspace carve
  uint8_t* wsp = (uint8_t*)d_ws; size_t off = 0;
  auto alloc = [&](size_t bytes) -> void* {
    void* r = wsp + off; off += (bytes + 255) & ~(size_t)255; return r;
  };
  _Float16* Abuf = (_Float16*)alloc((size_t)CHUNK * 1024 * 2);
  _Float16* Bbuf = (_Float16*)alloc((size_t)128 * 1024 * 2);
  float* srN  = (float*)alloc(4ull * 128 * 128 * 3 * 4);
  float* e1   = (float*)alloc(65536ull * 64 * 4);
  float* d1   = (float*)alloc(16384ull * 64 * 4);
  float* mm   = (float*)alloc(16384ull * 64 * 4);
  float* U    = (float*)alloc(65536ull * 64 * 4);
  float* uu   = (float*)alloc(65536ull * 64 * 4);
  float* uo   = U;  // reuse upsample buffer after 'up' conv consumes it
  float* srf  = (float*)alloc(16384ull * 64 * 4);
  float* F0   = (float*)alloc(16384ull * 64 * 4);
  float* G1   = (float*)alloc(4096ull * 64 * 4);
  float* G2   = (float*)alloc(4096ull * 64 * 4);
  float* G3   = (float*)alloc(1024ull * 64 * 4);
  float* inF  = (float*)alloc(1024ull * 64 * 4);
  _Float16* Axe  = (_Float16*)alloc(16384ull * 64 * 2);
  _Float16* Acat = (_Float16*)alloc(16384ull * 128 * 2);
  _Float16* Ag1  = (_Float16*)alloc(16384ull * 64 * 2);
  _Float16* Ag2  = (_Float16*)alloc(16384ull * 64 * 2);
  float* AGGb = (float*)alloc(16384ull * 64 * 4);
  float* Sa   = (float*)alloc(16384ull * 64 * 4);  // 16 slices x [1024][64]
  float* Sb   = (float*)alloc(16384ull * 64 * 4);
  float* T1   = (float*)alloc(16384ull * 64 * 4);
  float* Rb   = (float*)alloc(16384ull * 64 * 4);
  _Float16* Ah = (_Float16*)alloc(1024ull * 1024 * 2);
  _Float16* Th = (_Float16*)alloc(1024ull * 64 * 2);
  float* OUTR = (float*)alloc(16384ull * 128 * 4);

  // generic conv: NHWC f32 src -> GEMM (im2col chunked) -> epilogue
  auto conv = [&](const float* src, int Bn, int H, int Wd, int C,
                  int wIdx, int bIdx, const float* scale, const float* shift,
                  const float* resid, float* outF, _Float16* outH, int Cout,
                  int KH, int KW, int strd, int pad, int act) {
    int OH = (H + 2 * pad - KH) / strd + 1;
    int OW = (Wd + 2 * pad - KW) / strd + 1;
    int K = KH * KW * C, Kp = (K + 31) & ~31;
    int M = Bn * OH * OW;
    EW(k_wconv, (long long)Cout * Kp, F(wIdx), Bbuf, Cout, C, KH, KW, Kp);
    for (int m0 = 0; m0 < M; m0 += CHUNK) {
      int cnt = (M - m0 < CHUNK) ? (M - m0) : CHUNK;
      EW(k_im2col, (long long)cnt * Kp, src, Abuf, H, Wd, C, KH, KW, strd, pad,
         OH, OW, K, Kp, m0);
      k_gemm<<<dim3(cnt / 128, Cout / 64), 256, 0, stream>>>(
          Abuf, Bbuf, F(bIdx), scale, shift,
          resid ? resid + (size_t)m0 * Cout : nullptr, Cout,
          outF ? outF + (size_t)m0 * Cout : nullptr,
          outH ? outH + (size_t)m0 * Cout : nullptr, Cout, Kp, act);
    }
  };

  // linear: f16 A [M][Kp] @ w(K,N) + b
  auto lin = [&](const _Float16* A, int M, int K, int wIdx, int bIdx,
                 float* outF, _Float16* outH, int ldo, int N, int act) {
    int Kp = (K + 31) & ~31;  // == K for all our uses (64/128)
    EW(k_wlin, (long long)N * Kp, F(wIdx), Bbuf, K, N, Kp);
    k_gemm<<<dim3(M / 128, N / 64), 256, 0, stream>>>(
        A, Bbuf, F(bIdx), nullptr, nullptr, nullptr, 0, outF, outH, ldo, Kp, act);
  };

  // ---- unet ----
  EW(k_nchw2nhwc, 4ll * 128 * 128 * 3, sr, srN, 3, 128 * 128);
  conv(srN, 4, 128, 128, 3,  UNET + 3, UNET + 2, 0, 0, 0, e1, 0, 64, 3, 3, 1, 1, 1);
  conv(e1,  4, 128, 128, 64, UNET + 1, UNET + 0, 0, 0, 0, d1, 0, 64, 4, 4, 2, 1, 1);
  conv(d1,  4, 64, 64, 64,   UNET + 5, UNET + 4, 0, 0, 0, mm, 0, 64, 3, 3, 1, 1, 1);
  EW(k_upsample2, 65536ll * 64, mm, U, 64, 64, 64);
  conv(U,   4, 128, 128, 64, UNET + 9, UNET + 8, 0, 0, e1, uu, 0, 64, 3, 3, 1, 1, 1);
  conv(uu,  4, 128, 128, 64, UNET + 7, UNET + 6, 0, 0, 0, uo, 0, 64, 3, 3, 1, 1, 0);
  conv(uo,  4, 128, 128, 64, SRD + 1,  SRD + 0,  0, 0, 0, srf, 0, 64, 4, 4, 2, 1, 0);

  // ---- init feature chain ----
  conv(srf, 4, 64, 64, 64, INIT0 + 1, INIT0 + 0, 0, 0, 0, F0, 0, 64, 3, 3, 1, 1, 1);
  conv(F0,  4, 64, 64, 64, IDS + 1,  IDS + 0,  0, 0, 0, G1, 0, 64, 4, 4, 2, 1, 1);
  conv(G1,  4, 32, 32, 64, I1X1 + 1, I1X1 + 0, 0, 0, 0, G2, 0, 64, 1, 1, 1, 0, 0);
  conv(G2,  4, 32, 32, 64, IDS + 3,  IDS + 2,  0, 0, 0, G3, 0, 64, 4, 4, 2, 1, 1);
  conv(G3,  4, 16, 16, 64, I1X1 + 3, I1X1 + 2, 0, 0, 0, inF, 0, 64, 1, 1, 1, 0, 0);

  // ---- xe / agg MLPs ----
  EW(k_xe_gather, 16384ll * 64, xin, F(XE + 0), F(XE + 1), Axe);
  lin(Axe, 16384, 64, XE + 3, XE + 2, nullptr, Acat, 128, 64, 0);  // cols 0..63
  EW(k_srf_gather, 16384ll * 64, srf, Acat, 128, 64);              // cols 64..127
  lin(Acat, 16384, 128, AGG + 1, AGG + 0, nullptr, Ag1, 64, 64, 1);
  lin(Ag1,  16384, 64,  AGG + 3, AGG + 2, nullptr, Ag2, 64, 64, 1);
  lin(Ag2,  16384, 64,  AGG + 5, AGG + 4, AGGb, nullptr, 64, 64, 0);
  EW(k_build_state, 16ll * 1024 * 64, inF, AGGb, Sa);

  // ---- gate layers ----
  float* Sc = Sa; float* Sn = Sb;
  for (int g = 0; g < 10; g++) {
    int base = GATES + g * 192;
    for (int i = 0; i < 16; i++) {
      int l = i + 1;
      int hB = base + i * 4;        // h[i][0].b,w  h[i][1].b,w
      int wB = base + 64 + i * 8;   // bn1.b,g bn2.b,g c1.b,w c2.b,w
      // resblock (batched over all l slices): c1 + bn1 + elu
      conv(Sc, 4 * l, 16, 16, 64, wB + 5, wB + 4, F(wB + 1), F(wB + 0),
           nullptr, T1, 0, 64, 3, 3, 1, 1, 1);
      // c2 + bn2 + residual(x)
      conv(T1, 4 * l, 16, 16, 64, wB + 7, wB + 6, F(wB + 3), F(wB + 2),
           Sc, Rb, 0, 64, 3, 3, 1, 1, 0);
      // pack to h-conv K order (k = c*l + j)
      EW(k_pack_h, 1024ll * l * 64, Rb, Ah, l);
      // h0: 1x1 (l*64 -> 64), elu, f16 out
      EW(k_wconv, 64ll * l * 64, F(hB + 1), Bbuf, 64, l * 64, 1, 1, l * 64);
      k_gemm<<<dim3(8, 1), 256, 0, stream>>>(Ah, Bbuf, F(hB + 0), nullptr,
          nullptr, nullptr, 0, nullptr, Th, 64, l * 64, 1);
      // h1: 1x1 (64 -> 64) -> new state slice i
      EW(k_wconv, 64ll * 64, F(hB + 3), Bbuf, 64, 64, 1, 1, 64);
      k_gemm<<<dim3(8, 1), 256, 0, stream>>>(Th, Bbuf, F(hB + 2), nullptr,
          nullptr, nullptr, 0, Sn + (size_t)i * 1024 * 64, nullptr, 64, 64, 0);
    }
    float* t = Sc; Sc = Sn; Sn = t;
  }

  // ---- output MLP + fold ----
  EW(k_state_rows, 16384ll * 64, Sc, Acat, 128, 0);
  EW(k_srf_gather, 16384ll * 64, srf, Acat, 128, 64);
  lin(Acat, 16384, 128, POUT + 1, POUT + 0, nullptr, Ag1, 64, 64, 1);
  lin(Ag1,  16384, 64,  POUT + 3, POUT + 2, nullptr, Ag2, 64, 64, 1);
  lin(Ag2,  16384, 64,  POUT + 5, POUT + 4, OUTR, nullptr, 128, 128, 0);
  EW(k_fold_out, 4ll * 128 * 64 * 64, OUTR, (float*)d_out);
}
